// GatGraphClassifier_58231166599580
// MI455X (gfx1250) — matrix operand order repaired and checked by reading, not compiled, not run on hardware
//
#include <hip/hip_runtime.h>
#include <hip/hip_bf16.h>
#include <math.h>
#include <stdint.h>

// ---------- problem constants (match reference) ----------
#define GAT_B   4
#define GAT_N   1024
#define GAT_H   8
#define GAT_F   256
#define GAT_DIN 256
#define GAT_NC  10
#define GAT_BN  (GAT_B * GAT_N)    // 4096
#define GAT_HF  (GAT_H * GAT_F)    // 2048

typedef __bf16 bf16_t;
typedef __attribute__((ext_vector_type(2)))  __bf16 bf16x2;
typedef __attribute__((ext_vector_type(16))) __bf16 v16bf;
typedef __attribute__((ext_vector_type(8)))  float  v8f;
typedef unsigned int u32x4 __attribute__((ext_vector_type(4)));
typedef int          i32x8 __attribute__((ext_vector_type(8)));
typedef int          i32x4 __attribute__((ext_vector_type(4)));

#ifndef __has_builtin
#define __has_builtin(x) 0
#endif
#if __has_builtin(__builtin_amdgcn_tensor_load_to_lds) && \
    __has_builtin(__builtin_amdgcn_s_wait_tensorcnt)
#define GAT_HAVE_TDM 1
#else
#define GAT_HAVE_TDM 0
#endif

// LDS B-tile: 32 (K) x 128 (N) bf16, row stride padded to 136 elements (272B)
// so lane-indexed K-row reads land on distinct banks (bank = 4*lane mod 64).
#define LDSB_STRIDE 136

// ---------- fp32 -> bf16 convert ----------
__global__ void f2bf_kernel(const float* __restrict__ src, bf16_t* __restrict__ dst, long n) {
    long i = (long)blockIdx.x * blockDim.x + threadIdx.x;
    if (i < n) dst[i] = (bf16_t)src[i];
}

// ---------- WMMA bf16 GEMM, LDS-staged B (TDM when available) ----------
// C[M,Nc] (+)= A[M,K] * B[K,Nc]; A/B bf16 row-major; C fp32; optional bf16 shadow Cb.
// Batched via blockIdx.z: offA = z*aBatch ; b=z/zdiv, h=z%zdiv ; offBC = b*outer + h*inner.
// 256 threads = 8 waves arranged 4(M) x 2(N); wave tile 16x64 (4 WMMA sharing A);
// block tile 64(M) x 128(N). Grids must divide M/64, Nc/128, K/32 exactly (they do).
__global__ __launch_bounds__(256) void wmma_gemm_bf16(
    const bf16_t* __restrict__ A, const bf16_t* __restrict__ Bm,
    float* __restrict__ C, bf16_t* __restrict__ Cb,
    int M, int Nc, int K, int lda, int ldb, int ldc,
    int accumulate, int zdiv,
    long aBatch, long bOuter, long bInner, long cOuter, long cInner)
{
    __shared__ __align__(16) bf16_t ldsB[32 * LDSB_STRIDE];

    const int z  = blockIdx.z;
    const int zb = z / zdiv, zh = z % zdiv;
    A  += (long)z * aBatch;
    Bm += (long)zb * bOuter + (long)zh * bInner;
    const long cOff = (long)zb * cOuter + (long)zh * cInner;

    const int wave = threadIdx.x >> 5;
    const int lane = threadIdx.x & 31;
    const int wm = wave >> 1;              // 0..3 : M tile within block
    const int wn = wave & 1;               // 0..1 : 64-wide N slab within block
    const int tileM = blockIdx.y * 64 + wm * 16;
    const int nBase = blockIdx.x * 128;    // block N origin
    const int nWave = wn * 64;             // wave N origin inside LDS tile

    const int half = lane >> 4;            // lane group 0/1
    const int r    = lane & 15;            // A row within tile / C column

    v8f acc0 = {}, acc1 = {}, acc2 = {}, acc3 = {};

    for (int k0 = 0; k0 < K; k0 += 32) {
        __syncthreads();                   // previous iteration done with ldsB

        // ---- stage B tile (32 x 128 bf16) into LDS ----
#if GAT_HAVE_TDM
        if (wave == 0) {
            const uint64_t ga = (uint64_t)(uintptr_t)(Bm + (long)k0 * ldb + nBase);
            const uint32_t la = (uint32_t)(uintptr_t)&ldsB[0];
            u32x4 g0 = {};
            g0[0] = 1u;                                  // count=1 (valid user D#)
            g0[1] = la;                                  // lds_addr (bytes)
            g0[2] = (uint32_t)ga;                        // global_addr[31:0]
            g0[3] = (uint32_t)(ga >> 32) | 0x80000000u;  // global_addr[56:32] | type=2
            i32x8 g1 = {};
            // data_size=2B | pad_enable | pad_interval=64 dwords (code 5) | pad_amount=4 dwords (code 3)
            g1[0] = (int)0x07510000u;
            g1[1] = 0;                                   // tensor_dim0[15:0]=0 (dim0 = 1<<24)
            g1[2] = (int)0x00000100u;                    // tensor_dim0[31:16]=0x0100, tensor_dim1[15:0]=0
            g1[3] = (int)((128u << 16) | 0x0100u);       // tensor_dim1[31:16]=0x0100 | tile_dim0=128
            g1[4] = 32;                                  // tile_dim1=32, tile_dim2=0
            g1[5] = ldb;                                 // tensor_dim0_stride[31:0]
            g1[6] = 0;                                   // stride hi | tensor_dim1_stride lo
            g1[7] = 0;
            i32x4 g2 = {}, g3 = {};
            i32x8 g4 = {};
            __builtin_amdgcn_tensor_load_to_lds(g0, g1, g2, g3, g4, 0);
            __builtin_amdgcn_s_wait_tensorcnt(0);        // issuing wave drains TENSORcnt
        }
#else
        {
            const int t  = threadIdx.x;
            const int kr = t >> 3;                       // 0..31 K row
            const int ch = (t & 7) * 16;                 // 16-elem chunk in N
            const float4* s4 = (const float4*)(Bm + (long)(k0 + kr) * ldb + nBase + ch);
            float4* d4 = (float4*)&ldsB[kr * LDSB_STRIDE + ch];
            d4[0] = s4[0]; d4[1] = s4[1];
        }
#endif
        __syncthreads();                   // tile published to all waves

        // ---- A operand: lane holds row (tileM+r); vgpr v -> K pair per ISA 16-bit A layout ----
        v16bf a;
        const bf16_t* Arow = A + (long)(tileM + r) * lda + k0;
#pragma unroll
        for (int v = 0; v < 8; ++v) {
            const int kk = ((v < 4) ? (2 * v) : (16 + 2 * (v - 4))) + half * 8;
            bf16x2 p = *reinterpret_cast<const bf16x2*>(Arow + kk);
            a[2 * v] = p.x; a[2 * v + 1] = p.y;
        }
        if (k0 + 32 < K) __builtin_prefetch(Arow + 32, 0, 1);   // global_prefetch_b8

        // ---- 4 B operands from LDS; lane holds column r; vgpr v -> K pair same as A ----
#pragma unroll
        for (int s = 0; s < 4; ++s) {
            v16bf b;
            const bf16_t* brow = &ldsB[nWave + s * 16 + r];
#pragma unroll
            for (int v = 0; v < 8; ++v) {
                const int kk = ((v < 4) ? (2 * v) : (16 + 2 * (v - 4))) + half * 8;
                b[2 * v] = brow[kk * LDSB_STRIDE]; b[2 * v + 1] = brow[(kk + 1) * LDSB_STRIDE];
            }
            v8f cin = (s == 0) ? acc0 : (s == 1) ? acc1 : (s == 2) ? acc2 : acc3;
            v8f cout = __builtin_amdgcn_wmma_f32_16x16x32_bf16(
                false, a, false, b, (short)0, cin, false, false);
            if (s == 0) acc0 = cout; else if (s == 1) acc1 = cout;
            else if (s == 2) acc2 = cout; else acc3 = cout;
        }
    }

    // ---- store: C/D layout vgpr v -> row (v + 8*half), col r ----
    const int cm = tileM + half * 8;
#pragma unroll
    for (int s = 0; s < 4; ++s) {
        const int cn = nBase + nWave + s * 16 + r;
        v8f acc = (s == 0) ? acc0 : (s == 1) ? acc1 : (s == 2) ? acc2 : acc3;
#pragma unroll
        for (int v = 0; v < 8; ++v) {
            const long idx = cOff + (long)(cm + v) * ldc + cn;
            float val = acc[v];
            if (accumulate) val += C[idx];
            C[idx] = val;
            if (Cb) Cb[idx] = (bf16_t)val;
        }
    }
}

// ---------- attention score vectors: s_src/s_tgt[b,n,h] = <P[b,n,h,:], a[h,:]> ----------
__global__ void scores_kernel(const float* __restrict__ P,
                              const float* __restrict__ a_s, const float* __restrict__ a_t,
                              float* __restrict__ ssrc, float* __restrict__ stgt,
                              int BN, int H, int F) {
    const int gw   = (int)((blockIdx.x * (long)blockDim.x + threadIdx.x) >> 5);
    const int lane = threadIdx.x & 31;
    if (gw >= BN * H) return;
    const int bn = gw / H, h = gw % H;
    const float* p = P + (long)bn * H * F + (long)h * F;
    float s1 = 0.f, s2 = 0.f;
    for (int f = lane; f < F; f += 32) {
        const float x = p[f];
        s1 += x * a_s[h * F + f];
        s2 += x * a_t[h * F + f];
    }
    for (int o = 16; o > 0; o >>= 1) { s1 += __shfl_xor(s1, o); s2 += __shfl_xor(s2, o); }
    if (lane == 0) { ssrc[gw] = s1; stgt[gw] = s2; }
}

// ---------- softmax over neighbors j; attn stored bf16 as (B,H,N,N) ----------
__global__ __launch_bounds__(256) void attn_softmax_kernel(
    const float* __restrict__ ssrc, const float* __restrict__ stgt,
    const float* __restrict__ mask, bf16_t* __restrict__ attn, int N, int H)
{
    const int i = blockIdx.x, b = blockIdx.y, t = threadIdx.x;
    __shared__ float red[256];
    const float* mrow = mask + ((long)b * N + i) * N;

    for (int h = 0; h < H; ++h) {
        const float si = ssrc[((long)b * N + i) * H + h];
        float mx = -__builtin_inff();
        for (int j = t; j < N; j += 256) {
            float sc = si + stgt[((long)b * N + j) * H + h];
            sc = sc > 0.f ? sc : 0.2f * sc;
            sc += mrow[j];
            mx = fmaxf(mx, sc);
        }
        red[t] = mx; __syncthreads();
        for (int s = 128; s > 0; s >>= 1) { if (t < s) red[t] = fmaxf(red[t], red[t + s]); __syncthreads(); }
        mx = red[0]; __syncthreads();
        float sum = 0.f;
        for (int j = t; j < N; j += 256) {
            float sc = si + stgt[((long)b * N + j) * H + h];
            sc = sc > 0.f ? sc : 0.2f * sc;
            sc += mrow[j];
            sum += __expf(sc - mx);
        }
        red[t] = sum; __syncthreads();
        for (int s = 128; s > 0; s >>= 1) { if (t < s) red[t] += red[t + s]; __syncthreads(); }
        sum = red[0]; __syncthreads();
        const float inv = 1.f / sum;
        bf16_t* arow = attn + (((long)(b * H + h) * N + i) * N);
        for (int j = t; j < N; j += 256) {
            float sc = si + stgt[((long)b * N + j) * H + h];
            sc = sc > 0.f ? sc : 0.2f * sc;
            sc += mrow[j];
            arow[j] = (bf16_t)(__expf(sc - mx) * inv);
        }
        __syncthreads();
    }
}

// ---------- layer1 epilogue: ELU(Y + b1) in place + bf16 shadow ----------
__global__ void elu_bias_kernel(float* __restrict__ Y, bf16_t* __restrict__ Yb,
                                const float* __restrict__ bias, long n, int HF) {
    const long i = (long)blockIdx.x * blockDim.x + threadIdx.x;
    if (i >= n) return;
    const float v = Y[i] + bias[i % HF];
    const float e = v > 0.f ? v : (__expf(v) - 1.f);
    Y[i] = e; Yb[i] = (bf16_t)e;
}

// ---------- layer2 epilogue: mean over heads + bias ----------
__global__ void headmean_kernel(const float* __restrict__ Y, const float* __restrict__ b2,
                                float* __restrict__ X2, int BN, int H, int F) {
    const int i = blockIdx.x * blockDim.x + threadIdx.x;
    if (i >= BN * F) return;
    const int bn = i / F, f = i % F;
    const float* p = Y + (long)bn * H * F + f;
    float s = 0.f;
    for (int h = 0; h < H; ++h) s += p[(long)h * F];
    X2[i] = s * (1.f / H) + b2[f];
}

// ---------- mean pool over nodes ----------
__global__ void nodemean_kernel(const float* __restrict__ X2, float* __restrict__ g,
                                int B, int N, int F) {
    const int i = blockIdx.x * blockDim.x + threadIdx.x;
    if (i >= B * F) return;
    const int b = i / F, f = i % F;
    const float* p = X2 + (long)b * N * F + f;
    float s = 0.f;
    for (int n = 0; n < N; ++n) s += p[(long)n * F];
    g[i] = s * (1.f / N);
}

// ---------- final classifier ----------
__global__ void classifier_kernel(const float* __restrict__ g, const float* __restrict__ Wc,
                                  const float* __restrict__ bc, float* __restrict__ out,
                                  int B, int F, int NC) {
    const int i = threadIdx.x;
    if (i >= B * NC) return;
    const int b = i / NC, c = i % NC;
    float s = bc[c];
    for (int f = 0; f < F; ++f) s += g[b * F + f] * Wc[f * NC + c];
    out[i] = s;
}

// ---------- host side ----------
extern "C" void kernel_launch(void* const* d_in, const int* in_sizes, int n_in,
                              void* d_out, int out_size, void* d_ws, size_t ws_size,
                              hipStream_t stream) {
    (void)in_sizes; (void)n_in; (void)out_size; (void)ws_size;
    const float* features = (const float*)d_in[0];
    const float* mask     = (const float*)d_in[2];
    const float* W1       = (const float*)d_in[3];
    const float* a_src1   = (const float*)d_in[4];
    const float* a_tgt1   = (const float*)d_in[5];
    const float* skip1    = (const float*)d_in[6];
    const float* b1       = (const float*)d_in[7];
    const float* W2       = (const float*)d_in[8];
    const float* a_src2   = (const float*)d_in[9];
    const float* a_tgt2   = (const float*)d_in[10];
    const float* skip2    = (const float*)d_in[11];
    const float* b2       = (const float*)d_in[12];
    const float* Wc       = (const float*)d_in[13];
    const float* bc       = (const float*)d_in[14];
    float* out = (float*)d_out;

    char* ws = (char*)d_ws;
    size_t off = 0;
    auto carve = [&](size_t bytes) { void* p = ws + off; off += (bytes + 255) & ~(size_t)255; return p; };
    bf16_t* Xb   = (bf16_t*)carve((size_t)GAT_BN * GAT_DIN * 2);
    bf16_t* W1b  = (bf16_t*)carve((size_t)GAT_DIN * GAT_HF * 2);
    bf16_t* S1b  = (bf16_t*)carve((size_t)GAT_DIN * GAT_HF * 2);
    bf16_t* W2b  = (bf16_t*)carve((size_t)GAT_HF * GAT_HF * 2);
    bf16_t* S2b  = (bf16_t*)carve((size_t)GAT_HF * GAT_HF * 2);
    float*  P    = (float*) carve((size_t)GAT_BN * GAT_HF * 4);
    bf16_t* Pb   = (bf16_t*)carve((size_t)GAT_BN * GAT_HF * 2);
    float*  Y1   = (float*) carve((size_t)GAT_BN * GAT_HF * 4);
    bf16_t* X1b  = (bf16_t*)carve((size_t)GAT_BN * GAT_HF * 2);
    float*  Y2   = (float*) carve((size_t)GAT_BN * GAT_HF * 4);
    float*  X2   = (float*) carve((size_t)GAT_BN * GAT_F * 4);
    float*  ssrc = (float*) carve((size_t)GAT_BN * GAT_H * 4);
    float*  stgt = (float*) carve((size_t)GAT_BN * GAT_H * 4);
    bf16_t* attn = (bf16_t*)carve((size_t)GAT_B * GAT_H * GAT_N * GAT_N * 2);
    float*  g    = (float*) carve((size_t)GAT_B * GAT_F * 4);

    auto cvt = [&](const float* s, bf16_t* d, long n) {
        f2bf_kernel<<<(unsigned)((n + 255) / 256), 256, 0, stream>>>(s, d, n);
    };
    auto gemm = [&](const bf16_t* A, const bf16_t* Bm, float* C, bf16_t* Cb,
                    int M, int Nc, int K, int lda, int ldb, int ldc,
                    int accum, int batches, int zdiv,
                    long aB, long bO, long bI, long cO, long cI) {
        dim3 grid((unsigned)(Nc / 128), (unsigned)(M / 64), (unsigned)batches);
        wmma_gemm_bf16<<<grid, 256, 0, stream>>>(A, Bm, C, Cb, M, Nc, K, lda, ldb, ldc,
                                                 accum, zdiv, aB, bO, bI, cO, cI);
    };

    // bf16 copies of inputs/weights
    cvt(features, Xb,  (long)GAT_BN * GAT_DIN);
    cvt(W1,       W1b, (long)GAT_DIN * GAT_HF);
    cvt(skip1,    S1b, (long)GAT_DIN * GAT_HF);
    cvt(W2,       W2b, (long)GAT_HF * GAT_HF);
    cvt(skip2,    S2b, (long)GAT_HF * GAT_HF);

    const long agg_aB = (long)GAT_N * GAT_N;
    const long agg_o  = (long)GAT_N * GAT_HF;

    // ===== layer 1 =====
    gemm(Xb, S1b, Y1, nullptr, GAT_BN, GAT_HF, GAT_DIN, GAT_DIN, GAT_HF, GAT_HF, 0, 1, 1, 0,0,0,0,0);
    gemm(Xb, W1b, P,  Pb,      GAT_BN, GAT_HF, GAT_DIN, GAT_DIN, GAT_HF, GAT_HF, 0, 1, 1, 0,0,0,0,0);
    scores_kernel<<<(GAT_BN * GAT_H * 32) / 256, 256, 0, stream>>>(P, a_src1, a_tgt1, ssrc, stgt, GAT_BN, GAT_H, GAT_F);
    attn_softmax_kernel<<<dim3(GAT_N, GAT_B), 256, 0, stream>>>(ssrc, stgt, mask, attn, GAT_N, GAT_H);
    gemm(attn, Pb, Y1, nullptr, GAT_N, GAT_F, GAT_N, GAT_N, GAT_HF, GAT_HF,
         1, GAT_B * GAT_H, GAT_H, agg_aB, agg_o, GAT_F, agg_o, GAT_F);
    {
        const long n = (long)GAT_BN * GAT_HF;
        elu_bias_kernel<<<(unsigned)((n + 255) / 256), 256, 0, stream>>>(Y1, X1b, b1, n, GAT_HF);
    }

    // ===== layer 2 =====
    gemm(X1b, S2b, Y2, nullptr, GAT_BN, GAT_HF, GAT_HF, GAT_HF, GAT_HF, GAT_HF, 0, 1, 1, 0,0,0,0,0);
    gemm(X1b, W2b, P,  Pb,      GAT_BN, GAT_HF, GAT_HF, GAT_HF, GAT_HF, GAT_HF, 0, 1, 1, 0,0,0,0,0);
    scores_kernel<<<(GAT_BN * GAT_H * 32) / 256, 256, 0, stream>>>(P, a_src2, a_tgt2, ssrc, stgt, GAT_BN, GAT_H, GAT_F);
    attn_softmax_kernel<<<dim3(GAT_N, GAT_B), 256, 0, stream>>>(ssrc, stgt, mask, attn, GAT_N, GAT_H);
    gemm(attn, Pb, Y2, nullptr, GAT_N, GAT_F, GAT_N, GAT_N, GAT_HF, GAT_HF,
         1, GAT_B * GAT_H, GAT_H, agg_aB, agg_o, GAT_F, agg_o, GAT_F);
    headmean_kernel<<<(GAT_BN * GAT_F + 255) / 256, 256, 0, stream>>>(Y2, b2, X2, GAT_BN, GAT_H, GAT_F);

    // ===== readout =====
    nodemean_kernel<<<(GAT_B * GAT_F + 255) / 256, 256, 0, stream>>>(X2, g, GAT_B, GAT_N, GAT_F);
    classifier_kernel<<<1, 64, 0, stream>>>(g, Wc, bc, out, GAT_B, GAT_F, GAT_NC);
}
